// BinaryLinear_89833535963713
// MI455X (gfx1250) — matrix operand order repaired and checked by reading, not compiled, not run on hardware
//
#include <hip/hip_runtime.h>
#include <hip/hip_bf16.h>

typedef __attribute__((ext_vector_type(16))) _Float16 v16h;
typedef __attribute__((ext_vector_type(8)))  _Float16 v8h;
typedef __attribute__((ext_vector_type(4)))  _Float16 v4h;
typedef __attribute__((ext_vector_type(8)))  float    v8f;
typedef __attribute__((ext_vector_type(4)))  float    v4f;
typedef __attribute__((ext_vector_type(4)))  unsigned int v4u;
typedef __attribute__((ext_vector_type(2)))  unsigned int v2u;

#define M_DIM 8192
#define N_DIM 4096
#define K_DIM 4096
#define BM 128
#define BN 128
#define BK 32
#define THREADS 256   // 8 wave32s

// Out = X @ sign(W)^T + sign(b).  W,b come in as exactly +-1.0f (sign(randn)),
// so sign-bit extraction is exact and f16(+-1) is exact.
__global__ __launch_bounds__(THREADS)
__attribute__((amdgpu_waves_per_eu(2)))
void binlinear_wmma_f16(const float* __restrict__ X,
                        const float* __restrict__ W,
                        const float* __restrict__ Bias,
                        float* __restrict__ Out)
{
    __shared__ __align__(32) _Float16 As[2][BM * BK];  // x tile, f16
    __shared__ __align__(32) _Float16 Bs[2][BM * BK];  // sign(W) tile, f16 (row n, K-contig)

    const int tid  = threadIdx.x;
    const int n0   = blockIdx.x * BN;   // OUT dim
    const int m0   = blockIdx.y * BM;   // batch dim

    const int wave = tid >> 5;
    const int lane = tid & 31;
    const int wm   = (wave & 1) * 64;   // wave's M offset in block tile
    const int wn   = (wave >> 1) * 32;  // wave's N offset in block tile
    const int lr   = lane & 15;         // row/col within 16x16 tile
    const int hi   = lane >> 4;         // lane group (K-half select)

    // ---- global -> register staging: 4 x b128 per thread per matrix ----
    v4f ra[4], rb[4];

    auto load_tile = [&](int kt) {
        const float* xa = X + (size_t)m0 * K_DIM + (size_t)kt * BK;
        const float* wb = W + (size_t)n0 * K_DIM + (size_t)kt * BK;
#pragma unroll
        for (int i = 0; i < 4; ++i) {
            int id  = tid + THREADS * i;    // 0..1023 float4 slots
            int row = id >> 3;              // 8 float4 per 32-wide row
            int c4  = (id & 7) * 4;
            ra[i] = *(const v4f*)(xa + (size_t)row * K_DIM + c4);
            rb[i] = *(const v4f*)(wb + (size_t)row * K_DIM + c4);
        }
    };

    auto store_tile = [&](int buf) {
#pragma unroll
        for (int i = 0; i < 4; ++i) {
            int id  = tid + THREADS * i;
            int row = id >> 3;
            int c4  = (id & 7) * 4;
            // A: f32 -> f16 (v_cvt_pk_f16_f32 pairs)
            v4h ha;
#pragma unroll
            for (int j = 0; j < 4; ++j) ha[j] = (_Float16)ra[i][j];
            *(v4h*)(&As[buf][row * BK + c4]) = ha;
            // B: branchless binarize via sign bit: f16(+-1) = 0x3C00 | sign
            union { v4f f; v4u u; } w; w.f = rb[i];
            v2u hb;
            hb[0] = 0x3C003C00u | (w.u[1] & 0x80000000u) | ((w.u[0] & 0x80000000u) >> 16);
            hb[1] = 0x3C003C00u | (w.u[3] & 0x80000000u) | ((w.u[2] & 0x80000000u) >> 16);
            *(v2u*)(&Bs[buf][row * BK + c4]) = hb;
        }
    };

    // ---- accumulators: 4 (M) x 2 (N) tiles of 16x16 f32 ----
    v8f acc[4][2];
#pragma unroll
    for (int mi = 0; mi < 4; ++mi)
#pragma unroll
        for (int ni = 0; ni < 2; ++ni)
            acc[mi][ni] = (v8f){0.f, 0.f, 0.f, 0.f, 0.f, 0.f, 0.f, 0.f};

    union AF { v16h v; v8h h[2]; };

    auto compute = [&](int buf) {
        const _Float16* as = As[buf];
        const _Float16* bs = Bs[buf];
        // Front-load ALL fragments, then issue WMMAs back-to-back.
        v16h bfrag[2];
        AF   afrag[4];
#pragma unroll
        for (int ni = 0; ni < 2; ++ni) {
            int n = wn + ni * 16 + lr;                           // col n, K hi*16..hi*16+15
            bfrag[ni] = *(const v16h*)(bs + n * BK + hi * 16);   // 2x ds_load_b128
        }
#pragma unroll
        for (int mi = 0; mi < 4; ++mi) {
            int m = wm + mi * 16 + lr;                           // row m, K hi*16..hi*16+15
            afrag[mi].h[0] = *(const v8h*)(as + m * BK + hi * 16);
            afrag[mi].h[1] = *(const v8h*)(as + m * BK + hi * 16 + 8);
        }
#pragma unroll
        for (int mi = 0; mi < 4; ++mi)
#pragma unroll
            for (int ni = 0; ni < 2; ++ni)
                acc[mi][ni] = __builtin_amdgcn_wmma_f32_16x16x32_f16(
                    false, afrag[mi].v, false, bfrag[ni],
                    (short)0, acc[mi][ni], false, false);
    };

    // ---- main K loop: double-buffered LDS, phase-pinned software pipeline ----
    const int KT = K_DIM / BK;  // 128
    load_tile(0);
    store_tile(0);
    __syncthreads();
#pragma unroll 1
    for (int kt = 0; kt < KT - 1; ++kt) {
        load_tile(kt + 1);                  // issue 8 b128 loads (no waits here)
        __builtin_amdgcn_sched_barrier(0);  // keep loads above the WMMA block
        compute(kt & 1);                    // ds_loads + 8 WMMAs overlap load latency
        __builtin_amdgcn_sched_barrier(0);  // keep wait+cvt+store below compute
        store_tile((kt + 1) & 1);           // progressive s_wait_loadcnt, cvt, ds_store
        __syncthreads();
    }
    compute((KT - 1) & 1);

    // ---- epilogue: D layout (VGPR r: M = r + hi*8, N = lr), add sign(bias) ----
#pragma unroll
    for (int ni = 0; ni < 2; ++ni) {
        int gcol = n0 + wn + ni * 16 + lr;
        float bsgn = (Bias[gcol] >= 0.f) ? 1.f : -1.f;
#pragma unroll
        for (int mi = 0; mi < 4; ++mi) {
            int grow = m0 + wm + mi * 16 + hi * 8;
#pragma unroll
            for (int r = 0; r < 8; ++r) {
                Out[(size_t)(grow + r) * N_DIM + gcol] = acc[mi][ni][r] + bsgn;
            }
        }
    }
}

extern "C" void kernel_launch(void* const* d_in, const int* in_sizes, int n_in,
                              void* d_out, int out_size, void* d_ws, size_t ws_size,
                              hipStream_t stream) {
    (void)in_sizes; (void)n_in; (void)d_ws; (void)ws_size; (void)out_size;
    const float* X    = (const float*)d_in[0];  // [8192, 4096] f32
    const float* W    = (const float*)d_in[1];  // [4096, 4096] f32 (+-1)
    const float* Bias = (const float*)d_in[2];  // [4096] f32 (+-1)
    float* Out        = (float*)d_out;          // [8192, 4096] f32

    dim3 grid(N_DIM / BN, M_DIM / BM);  // (32, 64): N fastest -> X strip stays hot in L2
    dim3 block(THREADS);
    binlinear_wmma_f16<<<grid, block, 0, stream>>>(X, W, Bias, Out);
}
